// MyAGNN_60241211293935
// MI455X (gfx1250) — matrix operand order repaired and checked by reading, not compiled, not run on hardware
//
#include <hip/hip_runtime.h>
#include <hip/hip_bf16.h>

typedef __attribute__((ext_vector_type(16))) _Float16 v16h;
typedef __attribute__((ext_vector_type(8)))  float    v8f;

#define EPSV 1e-12f

// ---------------------------------------------------------------------------
// Float atomic-max via int/uint ordering trick (works for any sign, init=-inf)
// ---------------------------------------------------------------------------
__device__ __forceinline__ void atomicMaxFloat(float* addr, float val) {
    if (val >= 0.0f) {
        atomicMax((int*)addr, __float_as_int(val));
    } else {
        atomicMin((unsigned int*)addr, __float_as_uint(val));
    }
}

// ---------------------------------------------------------------------------
// Convert weight matrices to f16 for WMMA (tiny: 64x128 + 64x64)
// ---------------------------------------------------------------------------
__global__ void agnn_convert_weights(const float* __restrict__ W1,
                                     const float* __restrict__ W2,
                                     _Float16* __restrict__ W1h,
                                     _Float16* __restrict__ W2h) {
    int i = blockIdx.x * blockDim.x + threadIdx.x;
    if (i < 64 * 128) W1h[i] = (_Float16)W1[i];
    if (i < 64 * 64)  W2h[i] = (_Float16)W2[i];
}

// ---------------------------------------------------------------------------
// WMMA GEMM: out[N x 64] = act(A[N x K] @ Bw^T + bias), Bw is [64 x K] f16.
// blockDim = 128 (4 waves). Each wave: one 16x16 output tile, K/32 WMMA ops.
// A fragment layout (16-bit A 16x32): lanes 0-15 hold K {0..7,16..23},
// lanes 16-31 hold K {8..15,24..31}. B fragment: lanes 0-15 K {0..7,16..23},
// lanes 16-31 K {8..15,24..31}, N = lane&15. D: VGPR r -> M = r + 8*hi, N = lane&15.
// ---------------------------------------------------------------------------
template<int K, bool RELU>
__global__ void agnn_gemm_wmma(const float* __restrict__ A,
                               const _Float16* __restrict__ Bw,
                               const float* __restrict__ bias,
                               float* __restrict__ out, int N) {
    const int lane = threadIdx.x & 31;
    const int wave = threadIdx.x >> 5;          // 0..3 -> output col tile
    const int hi   = lane >> 4;                 // 0 or 1
    const int l15  = lane & 15;
    const int row0 = blockIdx.x * 16;

    int arow = row0 + l15;
    if (arow >= N) arow = N - 1;                // clamp (stores are guarded)

    v8f c = {};
#pragma unroll
    for (int k0 = 0; k0 < K; k0 += 32) {
        // ---- A fragment: row = l15, two runs of 8 consecutive K values ----
        v16h a;
        const float* ap1 = A + (size_t)arow * K + k0 + hi * 8;
        const float* ap2 = ap1 + 16;
#pragma unroll
        for (int i = 0; i < 8; ++i) {
            a[i]     = (_Float16)ap1[i];
            a[i + 8] = (_Float16)ap2[i];
        }
        // ---- B fragment: col n = wave*16 + l15, two runs of 8 consecutive K values ----
        v16h b;
        const _Float16* bp = Bw + (size_t)(wave * 16 + l15) * K + k0 + hi * 8;
#pragma unroll
        for (int i = 0; i < 8; ++i) { b[i] = bp[i]; b[i + 8] = bp[16 + i]; }

        c = __builtin_amdgcn_wmma_f32_16x16x32_f16(
                /*neg_a=*/false, a, /*neg_b=*/false, b,
                /*c_mod=*/(short)0, c, /*reuse_a=*/false, /*reuse_b=*/false);
    }

    const int   ncol = wave * 16 + l15;
    const float bv   = bias[ncol];
#pragma unroll
    for (int r = 0; r < 8; ++r) {
        int m    = r + hi * 8;
        int orow = row0 + m;
        if (orow < N) {
            float v = c[r] + bv;
            if (RELU) v = v > 0.0f ? v : 0.0f;
            out[(size_t)orow * 64 + ncol] = v;
        }
    }
}

// ---------------------------------------------------------------------------
// Per-layer init: zero accumulation buffer, smax = -inf, denom = 0
// ---------------------------------------------------------------------------
__global__ void agnn_init_layer(float* __restrict__ smax, float* __restrict__ denom,
                                float* __restrict__ hNext, int N) {
    int i = blockIdx.x * blockDim.x + threadIdx.x;
    if (i < N * 64) hNext[i] = 0.0f;
    if (i < N) {
        smax[i]  = -__builtin_huge_valf();
        denom[i] = 0.0f;
    }
}

// ---------------------------------------------------------------------------
// Row-normalize features: one wave32 per node (64 dims = 2 per lane)
// ---------------------------------------------------------------------------
__global__ void agnn_normalize(const float* __restrict__ h,
                               float* __restrict__ xn, int N) {
    int node = (blockIdx.x * blockDim.x + threadIdx.x) >> 5;
    int lane = threadIdx.x & 31;
    if (node >= N) return;
    float a = h[(size_t)node * 64 + lane];
    float b = h[(size_t)node * 64 + 32 + lane];
    float ss = a * a + b * b;
#pragma unroll
    for (int m = 16; m >= 1; m >>= 1) ss += __shfl_xor(ss, m, 32);
    float rn = 1.0f / (sqrtf(ss) + EPSV);
    xn[(size_t)node * 64 + lane]      = a * rn;
    xn[(size_t)node * 64 + 32 + lane] = b * rn;
}

// ---------------------------------------------------------------------------
// Edge score: one wave32 per edge; cos-sim via wave reduction; segment max
// ---------------------------------------------------------------------------
__global__ void agnn_edge_score(const float* __restrict__ xn,
                                const int* __restrict__ row,
                                const int* __restrict__ col,
                                float* __restrict__ esc,
                                float* __restrict__ smax, int E) {
    int e    = (blockIdx.x * blockDim.x + threadIdx.x) >> 5;
    int lane = threadIdx.x & 31;
    if (e >= E) return;
    int r = row[e], c = col[e];
    const float* pr = xn + (size_t)r * 64;
    const float* pc = xn + (size_t)c * 64;
    float s = pr[lane] * pc[lane] + pr[32 + lane] * pc[32 + lane];
#pragma unroll
    for (int m = 16; m >= 1; m >>= 1) s += __shfl_xor(s, m, 32);
    if (lane == 0) {
        esc[e] = s;
        atomicMaxFloat(&smax[r], s);
    }
}

// ---------------------------------------------------------------------------
// Stable-softmax numerator + segment sum (one thread per edge)
// ---------------------------------------------------------------------------
__global__ void agnn_edge_exp(float* __restrict__ esc,
                              const int* __restrict__ row,
                              const float* __restrict__ smax,
                              float* __restrict__ denom, int E) {
    int i = blockIdx.x * blockDim.x + threadIdx.x;
    if (i >= E) return;
    int r = row[i];
    float ev = __expf(esc[i] - smax[r]);
    esc[i] = ev;
    atomicAdd(&denom[r], ev);
}

// ---------------------------------------------------------------------------
// Aggregation: wave per CHUNK consecutive edges. Rows are sorted -> run-length
// accumulate in registers, atomics only on row change (~8x fewer atomics).
// ---------------------------------------------------------------------------
__global__ void agnn_aggregate(const float* __restrict__ h,
                               const int* __restrict__ row,
                               const int* __restrict__ col,
                               const float* __restrict__ ealpha,
                               const float* __restrict__ denom,
                               float* __restrict__ out, int E) {
    const int CHUNK = 8;
    int wid  = (blockIdx.x * blockDim.x + threadIdx.x) >> 5;
    int lane = threadIdx.x & 31;
    int e0   = wid * CHUNK;
    if (e0 >= E) return;
    int eEnd = e0 + CHUNK;
    if (eEnd > E) eEnd = E;

    float acc0 = 0.0f, acc1 = 0.0f;
    int curRow = row[e0];
    for (int e = e0; e < eEnd; ++e) {
        if (e + 1 < eEnd) {  // prefetch next source row (random gather)
            __builtin_prefetch(h + (size_t)col[e + 1] * 64, 0, 0);
        }
        int r = row[e];
        if (r != curRow) {
            atomicAdd(&out[(size_t)curRow * 64 + lane], acc0);
            atomicAdd(&out[(size_t)curRow * 64 + 32 + lane], acc1);
            acc0 = 0.0f; acc1 = 0.0f; curRow = r;
        }
        float coeff = ealpha[e] / fmaxf(denom[r], EPSV);
        const float* hc = h + (size_t)col[e] * 64;
        acc0 += coeff * hc[lane];
        acc1 += coeff * hc[32 + lane];
    }
    atomicAdd(&out[(size_t)curRow * 64 + lane], acc0);
    atomicAdd(&out[(size_t)curRow * 64 + 32 + lane], acc1);
}

// ---------------------------------------------------------------------------
// In-place ReLU
// ---------------------------------------------------------------------------
__global__ void agnn_relu(float* __restrict__ v, int n) {
    int i = blockIdx.x * blockDim.x + threadIdx.x;
    if (i < n) {
        float x = v[i];
        v[i] = x > 0.0f ? x : 0.0f;
    }
}

// ---------------------------------------------------------------------------
// Driver
// ---------------------------------------------------------------------------
extern "C" void kernel_launch(void* const* d_in, const int* in_sizes, int n_in,
                              void* d_out, int out_size, void* d_ws, size_t ws_size,
                              hipStream_t stream) {
    const float* x   = (const float*)d_in[0];
    const int*   row = (const int*)d_in[1];
    const int*   col = (const int*)d_in[2];
    const float* W1  = (const float*)d_in[3];
    const float* b1  = (const float*)d_in[4];
    const float* W2  = (const float*)d_in[5];
    const float* b2  = (const float*)d_in[6];

    const int N = in_sizes[0] / 128;   // IN_DIM = 128
    const int E = in_sizes[1];

    // Carve workspace (256B aligned slabs)
    char*  ws  = (char*)d_ws;
    size_t off = 0;
    auto alloc = [&](size_t bytes) -> void* {
        void* p = ws + off;
        off += (bytes + 255) & ~(size_t)255;
        return p;
    };
    float*    hA    = (float*)alloc((size_t)N * 64 * sizeof(float));
    float*    hB    = (float*)alloc((size_t)N * 64 * sizeof(float));
    float*    xn    = (float*)alloc((size_t)N * 64 * sizeof(float));
    float*    smax  = (float*)alloc((size_t)N * sizeof(float));
    float*    denom = (float*)alloc((size_t)N * sizeof(float));
    float*    esc   = (float*)alloc((size_t)E * sizeof(float));
    _Float16* W1h   = (_Float16*)alloc((size_t)64 * 128 * sizeof(_Float16));
    _Float16* W2h   = (_Float16*)alloc((size_t)64 * 64 * sizeof(_Float16));

    const int T = 256;

    agnn_convert_weights<<<(64 * 128 + T - 1) / T, T, 0, stream>>>(W1, W2, W1h, W2h);

    // h = relu(x @ W1^T + b1)  -- WMMA f16, f32 accumulate
    agnn_gemm_wmma<128, true><<<(N + 15) / 16, 128, 0, stream>>>(x, W1h, b1, hA, N);

    for (int t = 0; t < 4; ++t) {
        agnn_init_layer<<<((size_t)N * 64 + T - 1) / T, T, 0, stream>>>(smax, denom, hB, N);
        agnn_normalize<<<(N + 7) / 8, T, 0, stream>>>(hA, xn, N);                 // 8 waves/blk
        agnn_edge_score<<<(E + 7) / 8, T, 0, stream>>>(xn, row, col, esc, smax, E);
        agnn_edge_exp<<<(E + T - 1) / T, T, 0, stream>>>(esc, row, smax, denom, E);
        agnn_aggregate<<<(E + 63) / 64, T, 0, stream>>>(hA, row, col, esc, denom, hB, E);
        agnn_relu<<<((size_t)N * 64 + T - 1) / T, T, 0, stream>>>(hB, N * 64);
        float* tmp = hA; hA = hB; hB = tmp;
    }

    // out = h @ W2^T + b2
    agnn_gemm_wmma<64, false><<<(N + 15) / 16, 128, 0, stream>>>(hA, W2h, b2,
                                                                 (float*)d_out, N);
}
